// H2GCN_Net_21801253994546
// MI455X (gfx1250) — matrix operand order repaired and checked by reading, not compiled, not run on hardware
//
#include <hip/hip_runtime.h>
#include <hip/hip_bf16.h>

typedef __attribute__((ext_vector_type(16))) __bf16 v16bf;
typedef __attribute__((ext_vector_type(8)))  float  v8f;
typedef __attribute__((ext_vector_type(2)))  float  v2f;

#define NFEAT  1024
#define HIDDEN 64
#define NCLASS 16
#define RFIN   448   // (2^(K+1)-1)*HIDDEN, K=2

// ---------------------------------------------------------------------------
// K0: pack w_embed [1024,64] f32 (row-major, K-major) into bf16 WMMA B-fragments.
// Fragment f = (c*32 + kt): 32 lanes x 16 bf16. 16-bit B 32x16 layout:
// lane half h holds K in [8h,8h+8) then [16+8h,16+8h+8) (2 per VGPR).
// ---------------------------------------------------------------------------
__global__ void pack_wembed(const float* __restrict__ W, __bf16* __restrict__ Wb) {
    int t = blockIdx.x * blockDim.x + threadIdx.x;   // 32 kt * 4 c * 32 lanes
    if (t >= 32 * 4 * 32) return;
    int lane = t & 31;
    int kt   = (t >> 5) & 31;
    int c    = t >> 10;
    int n = (lane & 15) + c * 16;
    int h = lane >> 4;
    __bf16* dst = Wb + (((size_t)(c * 32 + kt) * 32 + lane) * 16);
#pragma unroll
    for (int v = 0; v < 8; ++v) {
        int k = kt * 32 + (v < 4 ? 8 * h + 2 * v : 16 + 8 * h + 2 * (v - 4));
        dst[2 * v + 0] = (__bf16)W[(size_t)k * HIDDEN + n];
        dst[2 * v + 1] = (__bf16)W[(size_t)(k + 1) * HIDDEN + n];
    }
}

// ---------------------------------------------------------------------------
// K1: r0 = relu(x @ w_embed)   [N,1024] x [1024,64] -> [N,64]
// 8 waves/block, 1 wave = one 16-row M-tile, 4 accumulators cover all 64 cols.
// A (16-bit 16x32): lane (m = l&15, h = l>>4) holds K in [8h,8h+8) and
// [16+8h,16+8h+8)  -> two 32B contiguous f32 chunks per lane, cvt to bf16.
// ---------------------------------------------------------------------------
__global__ void embed_gemm_relu(const float* __restrict__ X,
                                const __bf16* __restrict__ Wb,
                                float* __restrict__ r0, int nrows) {
    int lane = threadIdx.x & 31;
    int wave = threadIdx.x >> 5;
    int tile = blockIdx.x * 8 + wave;
    int ntiles = (nrows + 15) >> 4;
    if (tile >= ntiles) return;              // uniform per wave (EXEC all-1s for WMMA)
    int row0 = tile * 16;
    int m = lane & 15;
    int h = lane >> 4;
    int rowA = row0 + m; if (rowA > nrows - 1) rowA = nrows - 1;
    const float* px = X + (size_t)rowA * NFEAT + 8 * h;
    const v16bf* bbase = (const v16bf*)Wb + lane;

    v8f acc0 = {}, acc1 = {}, acc2 = {}, acc3 = {};
    for (int kt = 0; kt < NFEAT / 32; ++kt) {
        const float* p = px + kt * 32;
        float4 c0 = *(const float4*)(p);
        float4 c1 = *(const float4*)(p + 4);
        float4 c2 = *(const float4*)(p + 16);
        float4 c3 = *(const float4*)(p + 20);
        v16bf A;
        A[0]  = (__bf16)c0.x; A[1]  = (__bf16)c0.y; A[2]  = (__bf16)c0.z; A[3]  = (__bf16)c0.w;
        A[4]  = (__bf16)c1.x; A[5]  = (__bf16)c1.y; A[6]  = (__bf16)c1.z; A[7]  = (__bf16)c1.w;
        A[8]  = (__bf16)c2.x; A[9]  = (__bf16)c2.y; A[10] = (__bf16)c2.z; A[11] = (__bf16)c2.w;
        A[12] = (__bf16)c3.x; A[13] = (__bf16)c3.y; A[14] = (__bf16)c3.z; A[15] = (__bf16)c3.w;
        const v16bf* bp = bbase + (size_t)kt * 32;
        v16bf B0 = bp[0 * 32 * 32];
        v16bf B1 = bp[1 * 32 * 32];
        v16bf B2 = bp[2 * 32 * 32];
        v16bf B3 = bp[3 * 32 * 32];
        acc0 = __builtin_amdgcn_wmma_f32_16x16x32_bf16(false, A, false, B0, (short)0, acc0, false, false);
        acc1 = __builtin_amdgcn_wmma_f32_16x16x32_bf16(false, A, false, B1, (short)0, acc1, false, false);
        acc2 = __builtin_amdgcn_wmma_f32_16x16x32_bf16(false, A, false, B2, (short)0, acc2, false, false);
        acc3 = __builtin_amdgcn_wmma_f32_16x16x32_bf16(false, A, false, B3, (short)0, acc3, false, false);
    }
    // C/D layout: lanes 0-15 -> M=v, lanes 16-31 -> M=v+8; N = lane&15
#pragma unroll
    for (int v = 0; v < 8; ++v) {
        int r = row0 + v + 8 * h;
        if (r < nrows) {
            float* o = r0 + (size_t)r * HIDDEN + m;
            float f0 = acc0[v]; o[0]  = f0 > 0.f ? f0 : 0.f;
            float f1 = acc1[v]; o[16] = f1 > 0.f ? f1 : 0.f;
            float f2 = acc2[v]; o[32] = f2 > 0.f ? f2 : 0.f;
            float f3 = acc3[v]; o[48] = f3 > 0.f ? f3 : 0.f;
        }
    }
}

// ---------------------------------------------------------------------------
// K2: COO SpMM scatter:  out[rows[e], off:off+w] += val[e] * rin[cols[e], 0:w]
// (1<<lshift) lanes per edge, float4 gather per lane, f32 atomic scatter (L2).
// ---------------------------------------------------------------------------
__global__ void spmm_edges(const int* __restrict__ rows, const int* __restrict__ cols,
                           const float* __restrict__ vals,
                           const float* __restrict__ rin, int in_stride4,
                           float* __restrict__ out, int out_stride, int out_off,
                           int nedge, int lshift) {
    long long tid = (long long)blockIdx.x * blockDim.x + threadIdx.x;
    long long e = tid >> lshift;
    if (e >= nedge) return;
    int j = (int)(tid & ((1 << lshift) - 1));
    int r = rows[e];
    int c = cols[e];
    float v = vals[e];
    float4 d = ((const float4*)rin)[(long long)c * in_stride4 + j];
    float* o = out + (long long)r * out_stride + out_off + j * 4;
    atomicAdd(o + 0, v * d.x);
    atomicAdd(o + 1, v * d.y);
    atomicAdd(o + 2, v * d.z);
    atomicAdd(o + 3, v * d.w);
}

// ---------------------------------------------------------------------------
// K3: out = log_softmax([r0|rs1|rs2] @ Wc)  with f32 WMMA 16x16x4.
// Wc (448x16, 28KB) staged in LDS. A f32 16x4: lane(m,h): v0->K=2h, v1->K=2h+1.
// B f32 4x16 (mirror):  v0 -> rows {2h}, v1 -> rows {2h+1} at col m.
// ---------------------------------------------------------------------------
__global__ void classify_logsoftmax(const float* __restrict__ r0,
                                    const float* __restrict__ rs1,
                                    const float* __restrict__ rs2,
                                    const float* __restrict__ Wc,
                                    float* __restrict__ out, int nrows) {
    __shared__ float lWc[RFIN * NCLASS];
    for (int i = threadIdx.x; i < RFIN * NCLASS; i += blockDim.x) lWc[i] = Wc[i];
    __syncthreads();

    int lane = threadIdx.x & 31;
    int wave = threadIdx.x >> 5;
    int tile = blockIdx.x * 8 + wave;
    int ntiles = (nrows + 15) >> 4;
    if (tile >= ntiles) return;
    int row0 = tile * 16;
    int m = lane & 15;
    int h = lane >> 4;
    int rowA = row0 + m; if (rowA > nrows - 1) rowA = nrows - 1;

    v8f acc = {};
    for (int step = 0; step < RFIN / 4; ++step) {
        int k0 = step * 4;
        const float* src; int stride; int kl;
        if (k0 < HIDDEN)          { src = r0;  stride = HIDDEN;     kl = k0; }
        else if (k0 < 3 * HIDDEN) { src = rs1; stride = 2 * HIDDEN; kl = k0 - HIDDEN; }
        else                      { src = rs2; stride = 4 * HIDDEN; kl = k0 - 3 * HIDDEN; }
        const float* pa = src + (size_t)rowA * stride + kl + 2 * h;
        v2f A = { pa[0], pa[1] };
        v2f B = { lWc[(k0 + 2 * h) * NCLASS + m],
                  lWc[(k0 + 2 * h + 1) * NCLASS + m] };
        acc = __builtin_amdgcn_wmma_f32_16x16x4_f32(false, A, false, B, (short)0, acc, false, false);
    }

    // log_softmax over 16 classes: classes live across 16-lane half-wave
#pragma unroll
    for (int v = 0; v < 8; ++v) {
        float x = acc[v];
        float mx = x;
        mx = fmaxf(mx, __shfl_xor(mx, 1, 32));
        mx = fmaxf(mx, __shfl_xor(mx, 2, 32));
        mx = fmaxf(mx, __shfl_xor(mx, 4, 32));
        mx = fmaxf(mx, __shfl_xor(mx, 8, 32));
        float ex = expf(x - mx);
        float s = ex;
        s += __shfl_xor(s, 1, 32);
        s += __shfl_xor(s, 2, 32);
        s += __shfl_xor(s, 4, 32);
        s += __shfl_xor(s, 8, 32);
        int r = row0 + v + 8 * h;
        if (r < nrows) out[(size_t)r * NCLASS + m] = x - mx - logf(s);
    }
}

// ---------------------------------------------------------------------------
extern "C" void kernel_launch(void* const* d_in, const int* in_sizes, int n_in,
                              void* d_out, int out_size, void* d_ws, size_t ws_size,
                              hipStream_t stream) {
    const float* x      = (const float*)d_in[0];
    const int*   a1_idx = (const int*)d_in[1];
    const float* a1_val = (const float*)d_in[2];
    const int*   a2_idx = (const int*)d_in[3];
    const float* a2_val = (const float*)d_in[4];
    const float* w_e    = (const float*)d_in[5];
    const float* w_c    = (const float*)d_in[6];
    int N  = in_sizes[0] / NFEAT;
    int E1 = in_sizes[2];
    int E2 = in_sizes[4];

    // workspace carve-out
    char*   ws  = (char*)d_ws;
    __bf16* Wb  = (__bf16*)ws;                              // 128 KB of fragments
    float*  r0  = (float*)(ws + (128 << 10));               // [N,64]
    float*  rs1 = r0  + (size_t)N * HIDDEN;                 // [N,128]
    float*  rs2 = rs1 + (size_t)N * 2 * HIDDEN;             // [N,256]

    // accumulation targets must start at zero each call
    hipMemsetAsync(rs1, 0, (size_t)N * 2 * HIDDEN * sizeof(float), stream);
    hipMemsetAsync(rs2, 0, (size_t)N * 4 * HIDDEN * sizeof(float), stream);

    pack_wembed<<<16, 256, 0, stream>>>(w_e, Wb);

    int ntiles = (N + 15) / 16;
    int nblk   = (ntiles + 7) / 8;
    embed_gemm_relu<<<nblk, 256, 0, stream>>>(x, Wb, r0, N);

    // round 1: width 64 -> 16 lanes/edge   (rs1 = [A1@r0 | A2@r0])
    {
        long long t1 = (long long)E1 << 4;
        long long t2 = (long long)E2 << 4;
        spmm_edges<<<(int)((t1 + 255) / 256), 256, 0, stream>>>(
            a1_idx, a1_idx + E1, a1_val, r0, HIDDEN / 4, rs1, 2 * HIDDEN, 0, E1, 4);
        spmm_edges<<<(int)((t2 + 255) / 256), 256, 0, stream>>>(
            a2_idx, a2_idx + E2, a2_val, r0, HIDDEN / 4, rs1, 2 * HIDDEN, HIDDEN, E2, 4);
    }
    // round 2: width 128 -> 32 lanes/edge  (rs2 = [A1@rs1 | A2@rs1])
    {
        long long t1 = (long long)E1 << 5;
        long long t2 = (long long)E2 << 5;
        spmm_edges<<<(int)((t1 + 255) / 256), 256, 0, stream>>>(
            a1_idx, a1_idx + E1, a1_val, rs1, 2 * HIDDEN / 4, rs2, 4 * HIDDEN, 0, E1, 5);
        spmm_edges<<<(int)((t2 + 255) / 256), 256, 0, stream>>>(
            a2_idx, a2_idx + E2, a2_val, rs1, 2 * HIDDEN / 4, rs2, 4 * HIDDEN, 2 * HIDDEN, E2, 5);
    }

    classify_logsoftmax<<<nblk, 256, 0, stream>>>(r0, rs1, rs2, w_c, (float*)d_out, N);
}
